// GATEncoder_62319975465563
// MI455X (gfx1250) — matrix operand order repaired and checked by reading, not compiled, not run on hardware
//
#include <hip/hip_runtime.h>
#include <hip/hip_bf16.h>

typedef _Float16 f16;
typedef __attribute__((ext_vector_type(16))) _Float16 v16h;
typedef __attribute__((ext_vector_type(8)))  _Float16 v8h;
typedef __attribute__((ext_vector_type(8)))  float    v8f;

#define WMMA_F16(a, b, c) \
  __builtin_amdgcn_wmma_f32_16x16x32_f16(false, (a), false, (b), (short)0, (c), false, false)

// Force a (known-uniform) condition into an SGPR so the branch is scalar.
__device__ __forceinline__ bool uniform(bool c) {
  return __builtin_amdgcn_readfirstlane((int)c) != 0;
}

// ---------------------------------------------------------------------------
// B-fragment prep: bake a KxN(=64) f32 weight into WMMA-B lane order (f16).
// B 32x16: lanes 0-15 hold K=0..7,16..23 for column N=lane; lanes 16-31 hold K=8..15,24..31.
// ---------------------------------------------------------------------------
__global__ void prep_bfrag_kernel(const float* __restrict__ W, int KS, f16* __restrict__ out) {
  int tid = blockIdx.x * blockDim.x + threadIdx.x;
  int total = 4 * KS * 32;
  if (tid >= total) return;
  int lane = tid & 31;
  int ks   = (tid >> 5) % KS;
  int nb   = tid / (32 * KS);
  int half = lane >> 4;
  int col  = nb * 16 + (lane & 15);
  f16* o = out + (size_t)tid * 16;
#pragma unroll
  for (int i = 0; i < 16; ++i)
    o[i] = (f16)W[(size_t)(ks * 32 + (i < 8 ? half * 8 + i : 16 + half * 8 + (i - 8))) * 64 + col];
}

// ---------------------------------------------------------------------------
// edge_h = edge_attr[E,32] @ W_emb[32,64] + b_emb   (f16 out), plus column sums
// One wave per 16-edge tile (grid-stride). K=32 -> single WMMA k-step.
// ---------------------------------------------------------------------------
__global__ void __launch_bounds__(256) edge_embed_kernel(
    const float* __restrict__ attr, const float* __restrict__ b_emb,
    const f16* __restrict__ bfW, f16* __restrict__ edge_h,
    float* __restrict__ mean_acc, int E)
{
  __shared__ float lsum[64];
  const int tid = threadIdx.x;
  if (tid < 64) lsum[tid] = 0.0f;
  __syncthreads();

  const int lane = tid & 31;
  const int half = lane >> 4;
  const int lrow = lane & 15;
  const int gw = blockIdx.x * (blockDim.x >> 5) + (tid >> 5);
  const int nw = gridDim.x * (blockDim.x >> 5);

  v16h B[4];
#pragma unroll
  for (int nb = 0; nb < 4; ++nb)
    B[nb] = *(const v16h*)(bfW + (size_t)(nb * 32 + lane) * 16);

  float beC[4];
#pragma unroll
  for (int nb = 0; nb < 4; ++nb) beC[nb] = b_emb[nb * 16 + lrow];

  float csum[4] = {0.f, 0.f, 0.f, 0.f};
  const int ntiles = (E + 15) >> 4;
  for (int t = gw; t < ntiles; t += nw) {
    const bool full = uniform(t * 16 + 16 <= E);
    int arow = t * 16 + lrow; if (arow >= E) arow = E - 1;
    const float* ar = attr + (size_t)arow * 32;
    v16h a;
#pragma unroll
    for (int i = 0; i < 8; ++i) {
      a[i]     = (f16)ar[half * 8 + i];             // K = h*8 .. h*8+7
      a[8 + i] = (f16)ar[16 + half * 8 + i];        // K = 16+h*8 ..
    }
    f16* erow = edge_h + ((size_t)(t * 16 + half * 8)) * 64;
#pragma unroll
    for (int nb = 0; nb < 4; ++nb) {
      v8f c = {};
      c = WMMA_F16(a, B[nb], c);
      const int col = nb * 16 + lrow;
      if (full) {
#pragma unroll
        for (int r = 0; r < 8; ++r) {
          const float v = c[r] + beC[nb];
          erow[(size_t)r * 64 + col] = (f16)v;
          csum[nb] += v;
        }
      } else {
#pragma unroll
        for (int r = 0; r < 8; ++r) {
          const int orow = t * 16 + half * 8 + r;
          if (orow < E) {
            const float v = c[r] + beC[nb];
            edge_h[(size_t)orow * 64 + col] = (f16)v;
            csum[nb] += v;
          }
        }
      }
    }
  }
#pragma unroll
  for (int nb = 0; nb < 4; ++nb) atomicAdd(&lsum[nb * 16 + lrow], csum[nb]);
  __syncthreads();
  if (tid < 64) atomicAdd(&mean_acc[tid], lsum[tid]);
}

__global__ void finalize_mean_kernel(const float* __restrict__ acc, float invE,
                                     f16* __restrict__ mean_h) {
  const int c = threadIdx.x;
  if (c < 64) mean_h[c] = (f16)(acc[c] * invE);
}

// ---------------------------------------------------------------------------
// xout = h[N,64] @ W[64,64] + bias   (f32 out). One wave per 16-node tile.
// ---------------------------------------------------------------------------
__global__ void __launch_bounds__(256) node_xform_kernel(
    const float* __restrict__ h, const f16* __restrict__ bfW,
    const float* __restrict__ bias, float* __restrict__ xout, int Nn)
{
  const int tid = threadIdx.x;
  const int lane = tid & 31;
  const int half = lane >> 4;
  const int lrow = lane & 15;
  const int gw = blockIdx.x * (blockDim.x >> 5) + (tid >> 5);
  const int nw = gridDim.x * (blockDim.x >> 5);

  v16h B[4][2];
#pragma unroll
  for (int nb = 0; nb < 4; ++nb)
#pragma unroll
    for (int ks = 0; ks < 2; ++ks)
      B[nb][ks] = *(const v16h*)(bfW + (size_t)((nb * 2 + ks) * 32 + lane) * 16);

  float bC[4];
#pragma unroll
  for (int nb = 0; nb < 4; ++nb) bC[nb] = bias[nb * 16 + lrow];

  const int ntiles = (Nn + 15) >> 4;
  for (int t = gw; t < ntiles; t += nw) {
    const bool full = uniform(t * 16 + 16 <= Nn);
    int arow = t * 16 + lrow; if (arow >= Nn) arow = Nn - 1;
    const float* hr = h + (size_t)arow * 64;
    v16h a0, a1;
#pragma unroll
    for (int i = 0; i < 8; ++i) {
      a0[i]     = (f16)hr[half * 8 + i];
      a0[8 + i] = (f16)hr[16 + half * 8 + i];
      a1[i]     = (f16)hr[32 + half * 8 + i];
      a1[8 + i] = (f16)hr[48 + half * 8 + i];
    }
    float* orow = xout + ((size_t)(t * 16 + half * 8)) * 64;
#pragma unroll
    for (int nb = 0; nb < 4; ++nb) {
      v8f c = {};
      c = WMMA_F16(a0, B[nb][0], c);
      c = WMMA_F16(a1, B[nb][1], c);
      const int col = nb * 16 + lrow;
      if (full) {
#pragma unroll
        for (int r = 0; r < 8; ++r) orow[(size_t)r * 64 + col] = c[r] + bC[nb];
      } else {
#pragma unroll
        for (int r = 0; r < 8; ++r) {
          const int rw = t * 16 + half * 8 + r;
          if (rw < Nn) xout[(size_t)rw * 64 + col] = c[r] + bC[nb];
        }
      }
    }
  }
}

// ---------------------------------------------------------------------------
// Fused edge logits: e = edge_h@We (WMMA) + xl[src] + xr[dst]; LeakyReLU;
// s = e . att (cross-lane reduce); segment-max via order-preserving atomicMax.
// Rows >= E are the self loops (feature = mean_h, src=dst=node).
// ---------------------------------------------------------------------------
__global__ void __launch_bounds__(256) edge_logits_kernel(
    const f16* __restrict__ edge_h, const f16* __restrict__ mean_h,
    const int* __restrict__ eidx, const float* __restrict__ xl,
    const float* __restrict__ xr, const f16* __restrict__ bfWe,
    const float* __restrict__ att, float* __restrict__ s_arr,
    int* __restrict__ m_int, int E, int Etot)
{
  const int tid = threadIdx.x;
  const int lane = tid & 31;
  const int half = lane >> 4;
  const int lrow = lane & 15;
  const int gw = blockIdx.x * (blockDim.x >> 5) + (tid >> 5);
  const int nw = gridDim.x * (blockDim.x >> 5);

  v16h B[4][2];
#pragma unroll
  for (int nb = 0; nb < 4; ++nb)
#pragma unroll
    for (int ks = 0; ks < 2; ++ks)
      B[nb][ks] = *(const v16h*)(bfWe + (size_t)((nb * 2 + ks) * 32 + lane) * 16);

  float attc[4];
#pragma unroll
  for (int nb = 0; nb < 4; ++nb) attc[nb] = att[nb * 16 + lrow];

  const int ntiles = (Etot + 15) >> 4;
  for (int t = gw; t < ntiles; t += nw) {
    const int ebase = t * 16;
    const bool allReal = uniform(ebase + 16 <= E);
    const bool allLoop = uniform(ebase >= E);

    // --- A fragments (edge feature row for this lane), two 32-wide k-steps
    int arow = ebase + lrow; if (arow >= Etot) arow = Etot - 1;
    const f16* hr = (arow < E) ? (edge_h + (size_t)arow * 64) : mean_h;
    v8h p0 = *(const v8h*)(hr + half * 8);
    v8h p1 = *(const v8h*)(hr + 16 + half * 8);
    v8h p2 = *(const v8h*)(hr + 32 + half * 8);
    v8h p3 = *(const v8h*)(hr + 48 + half * 8);
    v16h a0 = __builtin_shufflevector(p0, p1, 0,1,2,3,4,5,6,7,8,9,10,11,12,13,14,15);
    v16h a1 = __builtin_shufflevector(p2, p3, 0,1,2,3,4,5,6,7,8,9,10,11,12,13,14,15);

    // --- src/dst of the 8 rows this lane owns in the D layout
    int srcs[8], dsts[8];
    if (allReal) {
      const int e0 = ebase + half * 8;
#pragma unroll
      for (int r = 0; r < 8; ++r) { srcs[r] = eidx[e0 + r]; dsts[r] = eidx[E + e0 + r]; }
    } else if (allLoop) {
      const int n0 = ebase - E + half * 8;
#pragma unroll
      for (int r = 0; r < 8; ++r) { srcs[r] = n0 + r; dsts[r] = n0 + r; }
    } else {
#pragma unroll
      for (int r = 0; r < 8; ++r) {
        int e = ebase + half * 8 + r; if (e >= Etot) e = Etot - 1;
        if (e < E) { srcs[r] = eidx[e]; dsts[r] = eidx[E + e]; }
        else       { srcs[r] = dsts[r] = e - E; }
      }
    }

    float ps[8] = {0.f,0.f,0.f,0.f,0.f,0.f,0.f,0.f};
#pragma unroll
    for (int nb = 0; nb < 4; ++nb) {
      v8f c = {};
      c = WMMA_F16(a0, B[nb][0], c);
      c = WMMA_F16(a1, B[nb][1], c);
      const int col = nb * 16 + lrow;
#pragma unroll
      for (int r = 0; r < 8; ++r) {
        float v = c[r] + xl[(size_t)srcs[r] * 64 + col] + xr[(size_t)dsts[r] * 64 + col];
        v = (v > 0.0f) ? v : 0.2f * v;              // LeakyReLU(0.2)
        ps[r] += v * attc[nb];
      }
    }
    // reduce across the 16 lanes of each half (columns)
#pragma unroll
    for (int m = 1; m < 16; m <<= 1)
#pragma unroll
      for (int r = 0; r < 8; ++r) ps[r] += __shfl_xor(ps[r], m, 32);

    // After reduction each lane of a half holds all 8 row sums; lanes 0-7 of
    // each half pick their own row via a cndmask chain and emit in parallel.
    float myv = ps[0];
    int mydst = dsts[0];
#pragma unroll
    for (int r = 1; r < 8; ++r) {
      myv   = (lrow == r) ? ps[r]   : myv;
      mydst = (lrow == r) ? dsts[r] : mydst;
    }
    const int mye = ebase + half * 8 + lrow;        // valid for lrow < 8
    if (lrow < 8 && mye < Etot) {
      s_arr[mye] = myv;
      int o = __float_as_int(myv);
      o = (o >= 0) ? o : (o ^ 0x7fffffff);          // order-preserving map
      atomicMax(&m_int[mydst], o);
    }
  }
}

// ---------------------------------------------------------------------------
// a = exp(s - m[dst]); denom[dst] += a; out[dst] += xl[src]*a. 64 lanes/edge.
// ---------------------------------------------------------------------------
__global__ void __launch_bounds__(256) scatter_kernel(
    const float* __restrict__ s_arr, const int* __restrict__ m_int,
    const int* __restrict__ eidx, const float* __restrict__ xl,
    float* __restrict__ outacc, float* __restrict__ denom, int E, int Etot)
{
  const int gid = blockIdx.x * blockDim.x + threadIdx.x;
  if (gid >= Etot * 64) return;
  const int c = gid & 63;
  const int e = gid >> 6;
  int srcn, dstn;
  if (e < E) { srcn = eidx[e]; dstn = eidx[E + e]; }
  else       { srcn = dstn = e - E; }
  const int mo = m_int[dstn];
  const float m = (mo >= 0) ? __int_as_float(mo) : __int_as_float(mo ^ 0x7fffffff);
  const float a = __expf(s_arr[e] - m);
  if (c == 0) atomicAdd(&denom[dstn], a);
  atomicAdd(&outacc[(size_t)dstn * 64 + c], xl[(size_t)srcn * 64 + c] * a);
}

__global__ void __launch_bounds__(256) normalize_kernel(
    const float* __restrict__ outacc, const float* __restrict__ denom,
    const float* __restrict__ bias, float* __restrict__ hout, int Nn, int relu)
{
  const int gid = blockIdx.x * blockDim.x + threadIdx.x;
  if (gid >= Nn * 64) return;
  const int n = gid >> 6, c = gid & 63;
  float v = outacc[gid] / (denom[n] + 1e-16f) + bias[c];
  if (relu) v = fmaxf(v, 0.0f);
  hout[gid] = v;
}

// ---------------------------------------------------------------------------
extern "C" void kernel_launch(void* const* d_in, const int* in_sizes, int n_in,
                              void* d_out, int out_size, void* d_ws, size_t ws_size,
                              hipStream_t stream)
{
  const float* x     = (const float*)d_in[0];
  const int*   eidx  = (const int*)d_in[1];
  const float* eattr = (const float*)d_in[2];
  const float* W_emb = (const float*)d_in[3];
  const float* b_emb = (const float*)d_in[4];
  const float* Wl    = (const float*)d_in[5];
  const float* bl    = (const float*)d_in[6];
  const float* Wr    = (const float*)d_in[7];
  const float* br    = (const float*)d_in[8];
  const float* We    = (const float*)d_in[9];
  const float* att   = (const float*)d_in[10];
  const float* bias  = (const float*)d_in[11];

  const int Nn   = in_sizes[0] / 64;
  const int E    = in_sizes[2] / 32;
  const int Etot = E + Nn;

  char* ws = (char*)d_ws;
  size_t off = 0;
  auto alloc = [&](size_t bytes) -> void* {
    void* p = ws + off;
    off = (off + bytes + 255) & ~(size_t)255;
    return p;
  };

  f16*   bf_emb   = (f16*)alloc((size_t)4 * 1 * 32 * 16 * sizeof(f16));
  f16*   bf_Wl    = (f16*)alloc((size_t)3 * 4 * 2 * 32 * 16 * sizeof(f16));
  f16*   bf_Wr    = (f16*)alloc((size_t)3 * 4 * 2 * 32 * 16 * sizeof(f16));
  f16*   bf_We    = (f16*)alloc((size_t)3 * 4 * 2 * 32 * 16 * sizeof(f16));
  float* mean_acc = (float*)alloc(64 * sizeof(float));
  f16*   mean_h   = (f16*)alloc(64 * sizeof(f16));
  f16*   edge_h   = (f16*)alloc((size_t)E * 64 * sizeof(f16));
  float* xl       = (float*)alloc((size_t)Nn * 64 * sizeof(float));
  float* xr       = (float*)alloc((size_t)Nn * 64 * sizeof(float));
  float* s_arr    = (float*)alloc((size_t)Etot * sizeof(float));
  int*   m_int    = (int*)alloc((size_t)Nn * sizeof(int));
  float* denom    = (float*)alloc((size_t)Nn * sizeof(float));
  float* outacc   = (float*)alloc((size_t)Nn * 64 * sizeof(float));
  float* hbuf     = (float*)alloc((size_t)Nn * 64 * sizeof(float));
  (void)ws_size; (void)n_in; (void)out_size;

  const size_t FRAG = (size_t)4 * 2 * 32 * 16;  // f16 elements per 64x64 weight

  hipMemsetAsync(mean_acc, 0, 64 * sizeof(float), stream);
  prep_bfrag_kernel<<<1, 256, 0, stream>>>(W_emb, 1, bf_emb);
  for (int l = 0; l < 3; ++l) {
    prep_bfrag_kernel<<<1, 256, 0, stream>>>(Wl + (size_t)l * 64 * 64, 2, bf_Wl + l * FRAG);
    prep_bfrag_kernel<<<1, 256, 0, stream>>>(Wr + (size_t)l * 64 * 64, 2, bf_Wr + l * FRAG);
    prep_bfrag_kernel<<<1, 256, 0, stream>>>(We + (size_t)l * 64 * 64, 2, bf_We + l * FRAG);
  }

  edge_embed_kernel<<<512, 256, 0, stream>>>(eattr, b_emb, bf_emb, edge_h, mean_acc, E);
  finalize_mean_kernel<<<1, 64, 0, stream>>>(mean_acc, 1.0f / (float)E, mean_h);

  for (int l = 0; l < 3; ++l) {
    const float* hcur = (l == 0) ? x : hbuf;
    node_xform_kernel<<<128, 256, 0, stream>>>(hcur, bf_Wl + l * FRAG, bl + l * 64, xl, Nn);
    node_xform_kernel<<<128, 256, 0, stream>>>(hcur, bf_Wr + l * FRAG, br + l * 64, xr, Nn);

    hipMemsetAsync(m_int, 0x80, (size_t)Nn * sizeof(int), stream);   // ~ -3.4e38 after decode
    hipMemsetAsync(denom, 0, (size_t)Nn * sizeof(float), stream);
    hipMemsetAsync(outacc, 0, (size_t)Nn * 64 * sizeof(float), stream);

    edge_logits_kernel<<<1024, 256, 0, stream>>>(edge_h, mean_h, eidx, xl, xr,
                                                 bf_We + l * FRAG, att + l * 64,
                                                 s_arr, m_int, E, Etot);

    const int total = Etot * 64;
    scatter_kernel<<<(total + 255) / 256, 256, 0, stream>>>(s_arr, m_int, eidx, xl,
                                                            outacc, denom, E, Etot);

    float* hout = (l == 2) ? (float*)d_out : hbuf;
    normalize_kernel<<<(Nn * 64 + 255) / 256, 256, 0, stream>>>(outacc, denom,
                                                                bias + l * 64, hout, Nn, l < 2);
  }
}
